// Transformer_12841952215655
// MI455X (gfx1250) — compile-verified, matrix operand order checked by reading, not run on hardware
//
#include <hip/hip_runtime.h>
#include <cstddef>

// ---------------- problem constants ----------------
constexpr int Bsz = 16, Seq = 77, Dm = 512, Ffn = 2048, Hh = 8, Dk = 64, Nl = 6, Vocab = 49408;
constexpr int Mrows = Bsz * Seq;                 // 1232 = 77 * 16  (exact 16-row WMMA tiles)
constexpr float EMB_SCALE = 22.62741699796952f;  // sqrt(512)

typedef __attribute__((ext_vector_type(16))) __bf16 v16bf;
typedef __attribute__((ext_vector_type(8)))  float  v8f;
typedef __attribute__((ext_vector_type(8)))  unsigned short v8us;

union FragBF { v16bf v; v8us h[2]; };

enum { GEMM_RELU = 1, GEMM_RES = 2 };

__device__ __forceinline__ unsigned short f2bf(float f) {
  unsigned u = __float_as_uint(f);
  u += 0x7FFFu + ((u >> 16) & 1u);               // round-to-nearest-even
  return (unsigned short)(u >> 16);
}

// ---------------- weight prep: W (KxN fp32) -> Wt (NxK bf16), LDS-tiled ----------------
__global__ __launch_bounds__(256) void transpose_to_bf16(
    const float* __restrict__ W, unsigned short* __restrict__ Wt, int K, int N)
{
  __shared__ float tile[32][33];
  const int n0 = blockIdx.x * 32, k0 = blockIdx.y * 32;
  const int tx = threadIdx.x, ty = threadIdx.y;   // 32 x 8
#pragma unroll
  for (int c = 0; c < 4; ++c)
    tile[ty + 8 * c][tx] = W[(size_t)(k0 + ty + 8 * c) * N + (n0 + tx)];
  __syncthreads();
#pragma unroll
  for (int c = 0; c < 4; ++c)
    Wt[(size_t)(n0 + ty + 8 * c) * K + (k0 + tx)] = f2bf(tile[tx][ty + 8 * c]);
}

// ---------------- WMMA GEMM: C[M,N] = act(A[M,K] @ W[K,N] + bias) (+res) ----------------
// A: M x K bf16 row-major.  Wt: N x K bf16 (pre-transposed).
// grid=(N/256, M/16), block=128 (4 waves). Each wave owns FOUR 16x16 C tiles (64 cols).
// The whole 16xK A panel is preloaded to LDS once (row stride K+8 -> conflict-free
// ds_load_b128), then the K loop is barrier-free and SOFTWARE-PIPELINED 2 deep:
// fragment loads for step k+32 are issued before the WMMAs of step k, in distinct
// registers, so the matrix pipe overlaps ~10 in-flight vector loads.
__global__ __launch_bounds__(128) void gemm_wmma_bf16(
    const unsigned short* __restrict__ A, const unsigned short* __restrict__ Wt,
    const float* __restrict__ bias, const float* __restrict__ res,
    float* __restrict__ C, unsigned short* __restrict__ Cb,
    int N, int K, int flags)
{
  extern __shared__ unsigned short As[];          // 16 x (K+8) bf16
  const int tid  = threadIdx.x;
  const int wave = tid >> 5;
  const int lane = tid & 31;
  const int r    = lane & 15;                     // row (A) / col (B,C) within tile
  const int kg   = lane >> 4;                     // K-group (ISA 7.12.2 lane split)
  const int kg8  = kg * 8;
  const int stride  = K + 8;
  const int rowBase = blockIdx.y << 4;
  const int colL    = (blockIdx.x << 8) + (wave << 6) + r;

  // ---- one-shot cooperative preload of the 16 x K A panel ----
  const int kc8 = K >> 3;                         // 16-byte chunks per row
  for (int c = tid; c < 2 * K; c += 128) {        // total chunks = 16*K/8
    const int row  = c / kc8;
    const int col8 = (c - row * kc8) << 3;
    *(v8us*)(As + row * stride + col8) =
        *(const v8us*)(A + (size_t)(rowBase + row) * K + col8);
  }
  __syncthreads();                                // the ONLY barrier in this kernel

  const unsigned short* wp0 = Wt + (size_t)(colL     ) * K;
  const unsigned short* wp1 = Wt + (size_t)(colL + 16) * K;
  const unsigned short* wp2 = Wt + (size_t)(colL + 32) * K;
  const unsigned short* wp3 = Wt + (size_t)(colL + 48) * K;
  const unsigned short* as  = As + r * stride + kg8;

  v8f a0 = {}, a1 = {}, a2 = {}, a3 = {};

  // Two independent fragment register sets (pipeline stages)
  FragBF pa, p0, p1, p2, p3;                      // stage P
  FragBF qa, q0, q1, q2, q3;                      // stage Q

#define LOAD_SET(FA, F0, F1, F2, F3, KO)                                        \
  {                                                                             \
    const int ko_ = (KO);                                                       \
    FA.h[0] = *(const v8us*)(as + ko_);                                         \
    FA.h[1] = *(const v8us*)(as + ko_ + 16);                                    \
    F0.h[0] = *(const v8us*)(wp0 + ko_ + kg8);                                  \
    F0.h[1] = *(const v8us*)(wp0 + ko_ + 16 + kg8);                             \
    F1.h[0] = *(const v8us*)(wp1 + ko_ + kg8);                                  \
    F1.h[1] = *(const v8us*)(wp1 + ko_ + 16 + kg8);                             \
    F2.h[0] = *(const v8us*)(wp2 + ko_ + kg8);                                  \
    F2.h[1] = *(const v8us*)(wp2 + ko_ + 16 + kg8);                             \
    F3.h[0] = *(const v8us*)(wp3 + ko_ + kg8);                                  \
    F3.h[1] = *(const v8us*)(wp3 + ko_ + 16 + kg8);                             \
  }
#define WMMA_SET(FA, F0, F1, F2, F3)                                                                     \
  a0 = __builtin_amdgcn_wmma_f32_16x16x32_bf16(false, FA.v, false, F0.v, (short)0, a0, false, false);    \
  a1 = __builtin_amdgcn_wmma_f32_16x16x32_bf16(false, FA.v, false, F1.v, (short)0, a1, false, false);    \
  a2 = __builtin_amdgcn_wmma_f32_16x16x32_bf16(false, FA.v, false, F2.v, (short)0, a2, false, false);    \
  a3 = __builtin_amdgcn_wmma_f32_16x16x32_bf16(false, FA.v, false, F3.v, (short)0, a3, false, false);

  LOAD_SET(pa, p0, p1, p2, p3, 0)                 // prologue: stage P holds k=0
  for (int k0 = 0; k0 < K; k0 += 64) {            // K is a multiple of 64
    LOAD_SET(qa, q0, q1, q2, q3, k0 + 32)         // stage Q loads fly over P's WMMAs
    if (k0 + 128 < K) {                           // keep B stream ~2 iterations ahead
      __builtin_prefetch(wp0 + k0 + 128, 0, 3);
      __builtin_prefetch(wp1 + k0 + 128, 0, 3);
      __builtin_prefetch(wp2 + k0 + 128, 0, 3);
      __builtin_prefetch(wp3 + k0 + 128, 0, 3);
    }
    WMMA_SET(pa, p0, p1, p2, p3)
    if (k0 + 64 < K) { LOAD_SET(pa, p0, p1, p2, p3, k0 + 64) }  // P loads fly over Q's WMMAs
    WMMA_SET(qa, q0, q1, q2, q3)
  }
#undef LOAD_SET
#undef WMMA_SET

  // ---- epilogue: C/D layout row = vgpr + 8*(lane>=16), col = lane&15 ----
#define GEMM_EPI(ACC, TOFF)                                                     \
  {                                                                             \
    const int col = colL + (TOFF);                                              \
    const float bv = bias[col];                                                 \
    _Pragma("unroll")                                                           \
    for (int vv = 0; vv < 8; ++vv) {                                            \
      const int row = rowBase + vv + (kg << 3);                                 \
      const size_t idx = (size_t)row * N + col;                                 \
      float cv = ACC[vv] + bv;                                                  \
      if (flags & GEMM_RELU) cv = fmaxf(cv, 0.f);                               \
      if (flags & GEMM_RES)  cv += res[idx];                                    \
      if (C)  C[idx]  = cv;                                                     \
      if (Cb) Cb[idx] = f2bf(cv);                                               \
    }                                                                           \
  }
  GEMM_EPI(a0, 0)
  GEMM_EPI(a1, 16)
  GEMM_EPI(a2, 32)
  GEMM_EPI(a3, 48)
#undef GEMM_EPI
}

// ---------------- embedding + positional encoding ----------------
__global__ __launch_bounds__(256) void embed_kernel(
    const int* __restrict__ tok, const float* __restrict__ emb,
    const float* __restrict__ pe, float* __restrict__ xo)
{
  const int t = blockIdx.x * 256 + threadIdx.x;   // exactly Mrows*Dm threads
  const int d = t & (Dm - 1);
  const int ms = t >> 9;
  const int s = ms % Seq;
  xo[t] = emb[(size_t)tok[ms] * Dm + d] * EMB_SCALE + pe[s * Dm + d];
}

// ---------------- LayerNorm: one wave per 512-wide row; dual fp32 + bf16 output --------
__global__ __launch_bounds__(256) void ln_kernel(
    const float* __restrict__ x, const float* __restrict__ g, const float* __restrict__ b,
    float* __restrict__ nout, unsigned short* __restrict__ nbf)
{
  const int wave = threadIdx.x >> 5, lane = threadIdx.x & 31;
  const int row = blockIdx.x * 8 + wave;
  const float* xr = x + (size_t)row * Dm;
  float4 xv[4];
  float s = 0.f, sq = 0.f;
#pragma unroll
  for (int c = 0; c < 4; ++c) {
    xv[c] = ((const float4*)xr)[lane + 32 * c];
    s  += xv[c].x + xv[c].y + xv[c].z + xv[c].w;
    sq += xv[c].x * xv[c].x + xv[c].y * xv[c].y + xv[c].z * xv[c].z + xv[c].w * xv[c].w;
  }
#pragma unroll
  for (int o = 16; o; o >>= 1) { s += __shfl_xor(s, o, 32); sq += __shfl_xor(sq, o, 32); }
  const float m  = s * (1.f / Dm);
  const float rs = rsqrtf(sq * (1.f / Dm) - m * m + 1e-6f);
#pragma unroll
  for (int c = 0; c < 4; ++c) {
    const int i4 = (lane + 32 * c) * 4;
    float4 ov;
    ov.x = (xv[c].x - m) * rs * g[i4 + 0] + b[i4 + 0];
    ov.y = (xv[c].y - m) * rs * g[i4 + 1] + b[i4 + 1];
    ov.z = (xv[c].z - m) * rs * g[i4 + 2] + b[i4 + 2];
    ov.w = (xv[c].w - m) * rs * g[i4 + 3] + b[i4 + 3];
    ((float4*)(nout + (size_t)row * Dm))[lane + 32 * c] = ov;
    uint2 pk;
    pk.x = (unsigned)f2bf(ov.x) | ((unsigned)f2bf(ov.y) << 16);
    pk.y = (unsigned)f2bf(ov.z) | ((unsigned)f2bf(ov.w) << 16);
    ((uint2*)(nbf + (size_t)row * Dm))[lane + 32 * c] = pk;
  }
}

// ---------------- attention (VALU; ~2% of total FLOPs) ----------------
constexpr int TOT_SCORES = Bsz * Hh * Seq * Seq;  // 758912

__global__ __launch_bounds__(256) void attn_scores(
    const float* __restrict__ q, const float* __restrict__ k,
    float* __restrict__ w, int causal)
{
  const int t = blockIdx.x * 256 + threadIdx.x;
  if (t >= TOT_SCORES) return;
  const int j = t % Seq; int rest = t / Seq;
  const int i = rest % Seq; rest /= Seq;
  const int h = rest % Hh;  const int b = rest / Hh;
  const float4* qp = (const float4*)(q + (size_t)(b * Seq + i) * Dm + h * Dk);
  const float4* kp = (const float4*)(k + (size_t)(b * Seq + j) * Dm + h * Dk);
  float acc = 0.f;
#pragma unroll
  for (int c = 0; c < Dk / 4; ++c) {
    float4 a = qp[c], bb = kp[c];
    acc += a.x * bb.x + a.y * bb.y + a.z * bb.z + a.w * bb.w;
  }
  float sv = acc * 0.125f;                        // 1/sqrt(64)
  if (causal && j > i) sv = -1e9f;
  sv = fminf(fmaxf(sv, -1e9f), 1e9f);
  w[(size_t)((b * Hh + h) * Seq + i) * Seq + j] = sv;
}

__global__ __launch_bounds__(128) void attn_softmax(float* __restrict__ w)
{
  const int wave = threadIdx.x >> 5, lane = threadIdx.x & 31;
  const int row = blockIdx.x * 4 + wave;          // Bsz*Hh*Seq rows, exact grid
  float* wr = w + (size_t)row * Seq;
  float v[3];
#pragma unroll
  for (int c = 0; c < 3; ++c) { const int j = lane + 32 * c; v[c] = (j < Seq) ? wr[j] : -3.0e38f; }
  float m = fmaxf(fmaxf(v[0], v[1]), v[2]);
#pragma unroll
  for (int o = 16; o; o >>= 1) m = fmaxf(m, __shfl_xor(m, o, 32));
  float e[3]; float sum = 0.f;
#pragma unroll
  for (int c = 0; c < 3; ++c) { const int j = lane + 32 * c; e[c] = (j < Seq) ? __expf(v[c] - m) : 0.f; sum += e[c]; }
#pragma unroll
  for (int o = 16; o; o >>= 1) sum += __shfl_xor(sum, o, 32);
  const float inv = 1.f / sum;
#pragma unroll
  for (int c = 0; c < 3; ++c) { const int j = lane + 32 * c; if (j < Seq) wr[j] = e[c] * inv; }
}

__global__ __launch_bounds__(256) void attn_av(
    const float* __restrict__ w, const float* __restrict__ v, unsigned short* __restrict__ obf)
{
  const int t = blockIdx.x * 256 + threadIdx.x;   // exactly Mrows*Dm threads
  const int d = t & (Dm - 1);
  const int ms = t >> 9;
  const int b = ms / Seq, i = ms % Seq;
  const int h = d >> 6;
  const float* wr = w + (size_t)((b * Hh + h) * Seq + i) * Seq;
  const float* vp = v + (size_t)(b * Seq) * Dm + d;
  float acc = 0.f;
  for (int j = 0; j < Seq; ++j) acc += wr[j] * vp[(size_t)j * Dm];
  obf[t] = f2bf(acc);
}

// ---------------- fake quantize (per-tensor int8 affine, eval forward) ----------------
__global__ __launch_bounds__(256) void fq_minmax(
    const float* __restrict__ x, int count, float* __restrict__ red)
{
  float mn = 3.0e38f, mx = -3.0e38f;
  for (int t = blockIdx.x * 256 + threadIdx.x; t < count; t += 256 * 256) {
    const float v = x[t]; mn = fminf(mn, v); mx = fmaxf(mx, v);
  }
  __shared__ float smn[256], smx[256];
  smn[threadIdx.x] = mn; smx[threadIdx.x] = mx; __syncthreads();
  for (int o = 128; o; o >>= 1) {
    if (threadIdx.x < o) {
      smn[threadIdx.x] = fminf(smn[threadIdx.x], smn[threadIdx.x + o]);
      smx[threadIdx.x] = fmaxf(smx[threadIdx.x], smx[threadIdx.x + o]);
    }
    __syncthreads();
  }
  if (threadIdx.x == 0) { red[blockIdx.x] = smn[0]; red[256 + blockIdx.x] = smx[0]; }
}

__global__ __launch_bounds__(256) void fq_finalize(const float* __restrict__ red, float* __restrict__ sc)
{
  __shared__ float smn[256], smx[256];
  smn[threadIdx.x] = red[threadIdx.x]; smx[threadIdx.x] = red[256 + threadIdx.x]; __syncthreads();
  for (int o = 128; o; o >>= 1) {
    if (threadIdx.x < o) {
      smn[threadIdx.x] = fminf(smn[threadIdx.x], smn[threadIdx.x + o]);
      smx[threadIdx.x] = fmaxf(smx[threadIdx.x], smx[threadIdx.x + o]);
    }
    __syncthreads();
  }
  if (threadIdx.x == 0) {
    const float mn = fminf(smn[0], 0.f), mx = fmaxf(smx[0], 0.f);
    const float scale = fmaxf((mx - mn) * (1.f / 254.f), 1e-12f);
    const float zp = fminf(fmaxf(rintf(-127.f - mn / scale), -127.f), 127.f);
    sc[0] = scale; sc[1] = zp;
  }
}

__global__ __launch_bounds__(256) void fq_apply(
    const float* __restrict__ x, const float* __restrict__ sc,
    unsigned short* __restrict__ outbf, int count)
{
  const int t = blockIdx.x * 256 + threadIdx.x;
  if (t >= count) return;
  const float scale = sc[0], zp = sc[1];
  const float qv = rintf(x[t] / scale) + zp;
  const float qc = fminf(fmaxf(qv, -127.f), 127.f);
  outbf[t] = f2bf((qc - zp) * scale);
}

// ---------------- host orchestration ----------------
extern "C" void kernel_launch(void* const* d_in, const int* in_sizes, int n_in,
                              void* d_out, int out_size, void* d_ws, size_t ws_size,
                              hipStream_t stream)
{
  (void)in_sizes; (void)n_in; (void)out_size; (void)ws_size;
  // Inputs flattened in setup_inputs() insertion order, params recursively:
  // 0:src 1:tgt 2:src_mask(all-False, unused) 3:tgt_mask(causal, computed in-kernel) 4:pe 5:emb
  // 6..21 enc{Wq,bq,Wk,bk,Wv,bv,Wo,bo,ln1g,ln1b,ln2g,ln2b,W1,b1,W2,b2}
  // 22..47 dec{sWq..sbo, cWq..cbo, ln1g..ln3b, W1,b1,W2,b2}
  // 48:enc_ng 49:enc_nb 50:dec_ng 51:dec_nb 52:out_W 53:out_b
  const int*   src = (const int*)d_in[0];
  const int*   tgt = (const int*)d_in[1];
  const float* pe  = (const float*)d_in[4];
  const float* emb = (const float*)d_in[5];
  auto F = [&](int i) { return (const float*)d_in[i]; };

  // ---- workspace (bump allocator, 256B aligned; ~83 MB total) ----
  char* wsp = (char*)d_ws;
  size_t off = 0;
  auto alloc = [&](size_t bytes) -> char* {
    char* p = wsp + off; off = (off + bytes + 255) & ~(size_t)255; return p;
  };
  unsigned short* Wt    = (unsigned short*)alloc((size_t)Vocab * Dm * 2);   // largest weight, reused
  unsigned short* Abf   = (unsigned short*)alloc((size_t)Mrows * Ffn * 2);
  unsigned short* Bbf   = (unsigned short*)alloc((size_t)Mrows * Ffn * 2);
  unsigned short* membf = (unsigned short*)alloc((size_t)Mrows * Dm * 2);
  float* x    = (float*)alloc((size_t)Mrows * Dm * 4);
  float* y    = (float*)alloc((size_t)Mrows * Dm * 4);
  float* memf = (float*)alloc((size_t)Mrows * Dm * 4);
  float* nbuf = (float*)alloc((size_t)Mrows * Dm * 4);
  float* q    = (float*)alloc((size_t)Mrows * Dm * 4);
  float* k    = (float*)alloc((size_t)Mrows * Dm * 4);
  float* v    = (float*)alloc((size_t)Mrows * Dm * 4);
  float* attw = (float*)alloc((size_t)TOT_SCORES * 4);
  float* red  = (float*)alloc(520 * 4);
  float* sc   = red + 512;

  const int MD_BLOCKS = Mrows * Dm / 256;         // 2464, exact

  auto gemm = [&](const unsigned short* A, const float* Wf, int K, int N,
                  const float* bias, const float* res, float* C, unsigned short* Cb, int flags) {
    transpose_to_bf16<<<dim3(N / 32, K / 32), dim3(32, 8), 0, stream>>>(Wf, Wt, K, N);
    const size_t shmem = (size_t)16 * (K + 8) * sizeof(unsigned short);
    gemm_wmma_bf16<<<dim3(N / 256, Mrows / 16), 128, shmem, stream>>>(A, Wt, bias, res, C, Cb, N, K, flags);
  };
  auto ln = [&](const float* xin, const float* g, const float* b, float* nout, unsigned short* nbf) {
    ln_kernel<<<dim3(Mrows / 8), 256, 0, stream>>>(xin, g, b, nout, nbf);
  };
  auto attn = [&](bool causal, unsigned short* obf) {
    attn_scores<<<dim3((TOT_SCORES + 255) / 256), 256, 0, stream>>>(q, k, attw, causal ? 1 : 0);
    attn_softmax<<<dim3(Bsz * Hh * Seq / 4), 128, 0, stream>>>(attw);
    attn_av<<<dim3(MD_BLOCKS), 256, 0, stream>>>(attw, v, obf);
  };
  auto fakequant = [&](const float* xin, unsigned short* outbf) {
    fq_minmax<<<dim3(256), 256, 0, stream>>>(xin, Mrows * Dm, red);
    fq_finalize<<<dim3(1), 256, 0, stream>>>(red, sc);
    fq_apply<<<dim3(MD_BLOCKS), 256, 0, stream>>>(xin, sc, outbf, Mrows * Dm);
  };

  // ================= encoder =================
  embed_kernel<<<dim3(MD_BLOCKS), 256, 0, stream>>>(src, emb, pe, x);
  for (int i = 0; i < Nl; ++i) {
    const size_t wD = (size_t)i * Dm * Dm, vD = (size_t)i * Dm;
    const size_t wF = (size_t)i * Dm * Ffn, vF = (size_t)i * Ffn;
    ln(x, F(14) + vD, F(15) + vD, nbuf, Abf);                                  // ln1 -> n (bf16 in Abf)
    gemm(Abf, F(6) + wD, Dm, Dm, F(7) + vD, nullptr, q, nullptr, 0);           // Q
    gemm(Abf, F(8) + wD, Dm, Dm, F(9) + vD, nullptr, k, nullptr, 0);           // K
    gemm(Abf, F(10) + wD, Dm, Dm, F(11) + vD, nullptr, v, nullptr, 0);         // V
    attn(false, Bbf);                                                          // o (bf16) -> Bbf
    gemm(Bbf, F(12) + wD, Dm, Dm, F(13) + vD, x, x, nullptr, GEMM_RES);        // x += o@Wo+bo
    ln(x, F(16) + vD, F(17) + vD, nbuf, Abf);                                  // ln2
    fakequant(nbuf, Abf);                                                      // fq(n) -> bf16 Abf
    gemm(Abf, F(18) + wF, Dm, Ffn, F(19) + vF, nullptr, nullptr, Bbf, GEMM_RELU); // h=relu(...)->Bbf bf16
    gemm(Bbf, F(20) + wF, Ffn, Dm, F(21) + vD, x, x, nullptr, GEMM_RES);       // x += h@W2+b2
  }
  ln(x, F(48), F(49), memf, membf);                                            // memory (fp32 + bf16)

  // ================= decoder =================
  embed_kernel<<<dim3(MD_BLOCKS), 256, 0, stream>>>(tgt, emb, pe, y);
  for (int i = 0; i < Nl; ++i) {
    const size_t wD = (size_t)i * Dm * Dm, vD = (size_t)i * Dm;
    const size_t wF = (size_t)i * Dm * Ffn, vF = (size_t)i * Ffn;
    // --- masked self-attention (causal) ---
    ln(y, F(38) + vD, F(39) + vD, nbuf, Abf);
    gemm(Abf, F(22) + wD, Dm, Dm, F(23) + vD, nullptr, q, nullptr, 0);
    gemm(Abf, F(24) + wD, Dm, Dm, F(25) + vD, nullptr, k, nullptr, 0);
    gemm(Abf, F(26) + wD, Dm, Dm, F(27) + vD, nullptr, v, nullptr, 0);
    attn(true, Bbf);
    gemm(Bbf, F(28) + wD, Dm, Dm, F(29) + vD, y, y, nullptr, GEMM_RES);
    // --- cross-attention (K,V from encoder memory) ---
    ln(y, F(40) + vD, F(41) + vD, nbuf, Abf);
    gemm(Abf,   F(30) + wD, Dm, Dm, F(31) + vD, nullptr, q, nullptr, 0);
    gemm(membf, F(32) + wD, Dm, Dm, F(33) + vD, nullptr, k, nullptr, 0);
    gemm(membf, F(34) + wD, Dm, Dm, F(35) + vD, nullptr, v, nullptr, 0);
    attn(false, Bbf);
    gemm(Bbf, F(36) + wD, Dm, Dm, F(37) + vD, y, y, nullptr, GEMM_RES);
    // --- FFN with fake-quant ---
    ln(y, F(42) + vD, F(43) + vD, nbuf, Abf);
    fakequant(nbuf, Abf);
    gemm(Abf, F(44) + wF, Dm, Ffn, F(45) + vF, nullptr, nullptr, Bbf, GEMM_RELU);
    gemm(Bbf, F(46) + wF, Ffn, Dm, F(47) + vD, y, y, nullptr, GEMM_RES);
  }

  // ================= final LN + vocab projection (dominant GEMM: 1232x512x49408) ========
  ln(y, F(50), F(51), nbuf, Abf);
  gemm(Abf, F(52), Dm, Vocab, F(53), nullptr, (float*)d_out, nullptr, 0);
}
